// RNN_72799695667281
// MI455X (gfx1250) — matrix operand order repaired and checked by reading, not compiled, not run on hardware
//
#include <hip/hip_runtime.h>

// ---- model constants -------------------------------------------------------
#define NU   20000     // unique words
#define TW   16        // chars per word
#define Bsz  64        // batch
#define Ssz  128       // words per sentence
#define Dd   256       // rnn cell dim
#define Asz  64        // attention size
#define CEe  100       // char embedding
#define WEe  100       // word embedding
#define KC   384       // padded K for char LSTM: 128 (x pad of 100) + 256 (h)
#define KSC  12        // KC/32
#define XPC  128       // x-part width (char)
#define KW   1408      // padded K for word LSTM: 1152 (x pad of 1124) + 256 (h)
#define KSW  44        // KW/32
#define XPW  1152      // x-part width (word)
#define NT   64        // 1024/16 output column tiles (4*D gates)

typedef _Float16 v8h  __attribute__((ext_vector_type(8)));
typedef _Float16 v16h __attribute__((ext_vector_type(16)));
typedef float    v8f  __attribute__((ext_vector_type(8)));

__device__ __forceinline__ float sigm(float x){ return 1.f/(1.f + expf(-x)); }

// ---- utility zero kernels --------------------------------------------------
__global__ void k_zero_f32(float* p, int n){ int i = blockIdx.x*256 + threadIdx.x; if (i < n) p[i] = 0.f; }
__global__ void k_zero_f16(_Float16* p, int n){ int i = blockIdx.x*256 + threadIdx.x; if (i < n) p[i] = (_Float16)0.f; }

// ---- pack f32 LSTM kernel weights into WMMA B-fragment layout (f16) --------
// pk[((nt*ksteps + ks)*32 + lane)*16 + i] = W[src_row(ks*32 + (lane>>4)*8 + (i&7) + (i>>3)*16)][nt*16 + (lane&15)]
__global__ void k_pack(const float* __restrict__ K, _Float16* __restrict__ pk,
                       int ksteps, int xpad, int nx){
  int idx = blockIdx.x*256 + threadIdx.x;
  int total = NT * ksteps * 512;
  if (idx >= total) return;
  int i    = idx & 15;
  int lane = (idx >> 4) & 31;
  int ks   = (idx >> 9) % ksteps;
  int nt   = (idx >> 9) / ksteps;
  int col  = nt*16 + (lane & 15);
  int krow = ks*32 + (lane >> 4)*8 + (i & 7) + (i >> 3)*16;     // padded-K row
  float v = 0.f;
  if (krow < xpad) { if (krow < nx) v = K[krow*1024 + col]; }          // x part (padded)
  else             { v = K[(nx + (krow - xpad))*1024 + col]; }         // h part
  pk[idx] = (_Float16)v;
}

// ---- char-LSTM: gather x_t (embedding lookup, fw/bw reversed) --------------
__global__ void k_char_x(const int* __restrict__ cs, const int* __restrict__ lens,
                         const float* __restrict__ emb, _Float16* __restrict__ xh,
                         int t, int bwd){
  int idx = blockIdx.x*256 + threadIdx.x;
  if (idx >= NU*XPC) return;
  int j = idx & (XPC-1);
  int m = idx >> 7;
  _Float16 v = (_Float16)0.f;
  if (j < CEe){
    int len = lens[m];
    int tt  = (bwd && t < len) ? (len - 1 - t) : t;
    int ch  = cs[m*TW + tt];
    v = (_Float16)emb[ch*CEe + j];
  }
  xh[(size_t)m*KC + j] = v;
}

// ---- fused LSTM step: z = [x|h] @ W + b via WMMA, then gates ---------------
// B fragments are addressed as ONE per-wave base pointer plus compile-time
// immediate offsets: (q*16*KSTEPS + ks)*512 halfs  (< 8M bytes -> fits the
// signed 24-bit IOFFSET of global_load_b128, so no per-load address VALU).
template<int KPAD, int KSTEPS, int KUNROLL>
__global__ void __launch_bounds__(256) k_lstm_step(
    const _Float16* __restrict__ xh_in, _Float16* __restrict__ xh_out,
    const _Float16* __restrict__ pkW, const float* __restrict__ bias,
    float* __restrict__ cwork, float* __restrict__ hwork,
    float* __restrict__ outbuf, const int* __restrict__ lens,
    int t, int bwd)
{
  constexpr int HOFF = KPAD - Dd;
  int wave  = (blockIdx.x*256 + threadIdx.x) >> 5;
  int lane  = threadIdx.x & 31;
  int mtile = wave >> 4;      // 16-row tile
  int nt    = wave & 15;      // gate-chunk column tile (within D=256)
  int col   = lane & 15;
  int hi    = lane >> 4;

  v8f acc[4];
  #pragma unroll
  for (int q = 0; q < 4; q++){
    float bv = bias[q*Dd + nt*16 + col];
    #pragma unroll
    for (int r = 0; r < 8; r++) acc[q][r] = bv;
  }

  // per-wave invariant base pointers; all loop offsets are immediates
  const _Float16* abase = xh_in + (size_t)(mtile*16 + col)*KPAD + hi*8;
  const _Float16* bbase = pkW + ((size_t)nt*KSTEPS*32 + lane)*16;

  #pragma unroll KUNROLL
  for (int ks = 0; ks < KSTEPS; ks++){
    v8h alo = *(const v8h*)(abase + ks*32);
    v8h ahi = *(const v8h*)(abase + ks*32 + 16);
    v16h a = __builtin_shufflevector(alo, ahi, 0,1,2,3,4,5,6,7,8,9,10,11,12,13,14,15);
    #pragma unroll
    for (int q = 0; q < 4; q++){
      v16h bfrag = *(const v16h*)(bbase + q*(16*KSTEPS*512) + ks*512);
      acc[q] = __builtin_amdgcn_wmma_f32_16x16x32_f16(
          false, a, false, bfrag, (short)0, acc[q], false, false);
    }
  }

  int nidx = nt*16 + col;
  #pragma unroll
  for (int r = 0; r < 8; r++){
    int m   = mtile*16 + r + hi*8;
    int len = lens[m];
    float iv = acc[0][r], gv = acc[1][r], fv = acc[2][r], ov = acc[3][r];
    float cold = cwork[(size_t)m*Dd + nidx];
    float cn = sigm(fv + 1.f)*cold + sigm(iv)*tanhf(gv);
    float hn = sigm(ov)*tanhf(cn);
    bool msk = (t < len);
    float hold = (float)xh_in[(size_t)m*KPAD + HOFF + nidx];
    if (msk){
      cwork[(size_t)m*Dd + nidx] = cn;
      if (hwork)  hwork[(size_t)m*Dd + nidx] = hn;
      if (outbuf){
        int so = bwd ? (len - 1 - t) : t;
        outbuf[((size_t)m*Ssz + so)*Dd + nidx] = hn;
      }
    }
    xh_out[(size_t)m*KPAD + HOFF + nidx] = (_Float16)(msk ? hn : hold);
  }
}

// ---- dump final char-LSTM state (c f32, h f16) into input_chars ------------
__global__ void k_char_fin(const float* __restrict__ cwork, const _Float16* __restrict__ xh,
                           _Float16* __restrict__ ic, int base){
  int idx = blockIdx.x*256 + threadIdx.x;
  if (idx >= NU*Dd) return;
  int n = idx & 255; int m = idx >> 8;
  ic[(size_t)m*1024 + base + n]      = (_Float16)cwork[idx];
  ic[(size_t)m*1024 + base + Dd + n] = xh[(size_t)m*KC + XPC + n];
}

// ---- build word-LSTM input features [B,S,XPW] (f16) ------------------------
__global__ void k_build_inp(const _Float16* __restrict__ ic, const float* __restrict__ wemb,
                            const int* __restrict__ cids, const int* __restrict__ wids,
                            _Float16* __restrict__ inp){
  int idx = blockIdx.x*256 + threadIdx.x;
  if (idx >= Bsz*Ssz*XPW) return;
  int j  = idx % XPW;
  int bs = idx / XPW;
  _Float16 v = (_Float16)0.f;
  if (j < 1024)            v = ic[(size_t)cids[bs]*1024 + j];
  else if (j < 1024 + WEe) v = (_Float16)wemb[(size_t)wids[bs]*WEe + (j - 1024)];
  inp[idx] = v;
}

// ---- word-LSTM: gather x_t (possibly sequence-reversed) --------------------
__global__ void k_word_x(const _Float16* __restrict__ inp, const int* __restrict__ lens,
                         _Float16* __restrict__ xh, int t, int bwd){
  int idx = blockIdx.x*256 + threadIdx.x;
  if (idx >= Bsz*XPW) return;
  int j = idx % XPW;
  int b = idx / XPW;
  int len = lens[b];
  int s = (bwd && t < len) ? (len - 1 - t) : t;
  xh[(size_t)b*KW + j] = inp[((size_t)b*Ssz + s)*XPW + j];
}

// ---- Bahdanau attention: ctx[b] = softmax(v.tanh(Wm*mem + Wq*q)) @ mem -----
__global__ void __launch_bounds__(256) k_attn(const float* __restrict__ mem,
    const int* __restrict__ lens, const float* __restrict__ hq,
    const float* __restrict__ Wm, const float* __restrict__ Wq,
    const float* __restrict__ vv, float* __restrict__ ctx)
{
  int b = blockIdx.x, tid = threadIdx.x;
  __shared__ float q[Asz];
  __shared__ float sc[Ssz];
  if (tid < Asz){
    float s = 0.f;
    for (int k = 0; k < Dd; k++) s += hq[b*Dd + k]*Wq[k*Asz + tid];
    q[tid] = s;
  }
  __syncthreads();
  if (tid < Ssz){
    const float* mrow = mem + ((size_t)b*Ssz + tid)*Dd;
    float sum = 0.f;
    for (int a = 0; a < Asz; a++){
      float key = 0.f;
      for (int k = 0; k < Dd; k++) key += mrow[k]*Wm[k*Asz + a];
      sum += vv[a]*tanhf(key + q[a]);
    }
    sc[tid] = sum;
  }
  __syncthreads();
  if (tid == 0){
    int len = lens[b];
    float mx = -1e30f;
    for (int s = 0; s < Ssz; s++) if (s < len && sc[s] > mx) mx = sc[s];
    float se = 0.f;
    for (int s = 0; s < Ssz; s++){ float e = (s < len) ? expf(sc[s]-mx) : 0.f; sc[s] = e; se += e; }
    float inv = 1.f/se;
    for (int s = 0; s < Ssz; s++) sc[s] *= inv;
  }
  __syncthreads();
  for (int n = tid; n < Dd; n += 256){
    float s = 0.f;
    for (int tt = 0; tt < Ssz; tt++) s += sc[tt]*mem[((size_t)b*Ssz + tt)*Dd + n];
    ctx[b*Dd + n] = s;
  }
}

// ---- final head: leaky_relu(states@W1+b1)@W2+b2 ----------------------------
__global__ void __launch_bounds__(256) k_head(const float* ctxf, const float* ctxb,
    const float* cf, const float* hf, const float* cb, const float* hb,
    const float* __restrict__ W1, const float* __restrict__ b1,
    const float* __restrict__ W2, const float* __restrict__ b2, float* out)
{
  int b = blockIdx.x, j = threadIdx.x;
  __shared__ float st[6*Dd];
  __shared__ float hid[256];
  for (int k = j; k < 6*Dd; k += 256){
    int seg = k >> 8, n = k & 255;
    const float* src = (seg==0)?ctxf:(seg==1)?ctxb:(seg==2)?cf:(seg==3)?hf:(seg==4)?cb:hb;
    st[k] = src[b*Dd + n];
  }
  __syncthreads();
  float h = b1[j];
  for (int k = 0; k < 6*Dd; k++) h += st[k]*W1[k*256 + j];
  hid[j] = (h > 0.f) ? h : 0.2f*h;
  __syncthreads();
  if (j < 2){
    float s = b2[j];
    for (int k = 0; k < 256; k++) s += hid[k]*W2[k*2 + j];
    out[b*2 + j] = s;
  }
}

// ---------------------------------------------------------------------------
extern "C" void kernel_launch(void* const* d_in, const int* in_sizes, int n_in,
                              void* d_out, int out_size, void* d_ws, size_t ws_size,
                              hipStream_t stream)
{
  (void)in_sizes; (void)n_in; (void)out_size; (void)ws_size;
  const int*   cs    = (const int*)d_in[0];
  const int*   clens = (const int*)d_in[1];
  const int*   cids  = (const int*)d_in[2];
  const int*   wids  = (const int*)d_in[3];
  const int*   slens = (const int*)d_in[4];
  const float* cemb  = (const float*)d_in[5];
  const float* wemb  = (const float*)d_in[6];
  const float* Kc[2]  = {(const float*)d_in[7],  (const float*)d_in[9]};
  const float* bcb[2] = {(const float*)d_in[8],  (const float*)d_in[10]};
  const float* Kw[2]  = {(const float*)d_in[11], (const float*)d_in[13]};
  const float* bwb[2] = {(const float*)d_in[12], (const float*)d_in[14]};
  const float* Wm[2]  = {(const float*)d_in[15], (const float*)d_in[18]};
  const float* Wq[2]  = {(const float*)d_in[16], (const float*)d_in[19]};
  const float* vv[2]  = {(const float*)d_in[17], (const float*)d_in[20]};
  const float* W1 = (const float*)d_in[21];
  const float* b1 = (const float*)d_in[22];
  const float* W2 = (const float*)d_in[23];
  const float* b2 = (const float*)d_in[24];

  // ---- workspace carve-up (all regions fully re-initialized every call) ----
  char* ws = (char*)d_ws;
  size_t off = 0;
  auto alloc = [&](size_t bytes)->char*{
    char* p = ws + off; off += (bytes + 255) & ~(size_t)255; return p;
  };
  _Float16* pkKc[2] = {(_Float16*)alloc((size_t)NT*KSC*512*2), (_Float16*)alloc((size_t)NT*KSC*512*2)};
  _Float16* pkKw[2] = {(_Float16*)alloc((size_t)NT*KSW*512*2), (_Float16*)alloc((size_t)NT*KSW*512*2)};
  _Float16* xhc[2]  = {(_Float16*)alloc((size_t)NU*KC*2),      (_Float16*)alloc((size_t)NU*KC*2)};
  float*    cworkc  = (float*)alloc((size_t)NU*Dd*4);
  _Float16* ic      = (_Float16*)alloc((size_t)NU*1024*2);
  _Float16* inpw    = (_Float16*)alloc((size_t)Bsz*Ssz*XPW*2);
  _Float16* xhw[2]  = {(_Float16*)alloc((size_t)Bsz*KW*2),     (_Float16*)alloc((size_t)Bsz*KW*2)};
  float*    cword[2]= {(float*)alloc((size_t)Bsz*Dd*4), (float*)alloc((size_t)Bsz*Dd*4)};
  float*    hword[2]= {(float*)alloc((size_t)Bsz*Dd*4), (float*)alloc((size_t)Bsz*Dd*4)};
  float*    outw[2] = {(float*)alloc((size_t)Bsz*Ssz*Dd*4), (float*)alloc((size_t)Bsz*Ssz*Dd*4)};
  float*    ctx[2]  = {(float*)alloc((size_t)Bsz*Dd*4), (float*)alloc((size_t)Bsz*Dd*4)};

  // ---- pack weights into WMMA fragment layout ------------------------------
  {
    int totC = NT*KSC*512, totW = NT*KSW*512;
    k_pack<<<(totC+255)/256,256,0,stream>>>(Kc[0], pkKc[0], KSC, XPC, CEe);
    k_pack<<<(totC+255)/256,256,0,stream>>>(Kc[1], pkKc[1], KSC, XPC, CEe);
    k_pack<<<(totW+255)/256,256,0,stream>>>(Kw[0], pkKw[0], KSW, XPW, 1124);
    k_pack<<<(totW+255)/256,256,0,stream>>>(Kw[1], pkKw[1], KSW, XPW, 1124);
  }

  // ---- char-level bi-LSTM over 20000 unique words --------------------------
  for (int dir = 0; dir < 2; dir++){
    int nxh = NU*KC;
    k_zero_f16<<<(nxh+255)/256,256,0,stream>>>(xhc[0], nxh);
    k_zero_f16<<<(nxh+255)/256,256,0,stream>>>(xhc[1], nxh);
    k_zero_f32<<<(NU*Dd+255)/256,256,0,stream>>>(cworkc, NU*Dd);
    for (int t = 0; t < TW; t++){
      k_char_x<<<(NU*XPC+255)/256,256,0,stream>>>(cs, clens, cemb, xhc[t&1], t, dir);
      k_lstm_step<KC,KSC,KSC><<<NU/8,256,0,stream>>>(
          xhc[t&1], xhc[(t+1)&1], pkKc[dir], bcb[dir],
          cworkc, nullptr, nullptr, clens, t, dir);
    }
    // after 16 steps, final h lives in xhc[0]
    k_char_fin<<<(NU*Dd+255)/256,256,0,stream>>>(cworkc, xhc[0], ic, dir*512);
  }

  // ---- word-level inputs ----------------------------------------------------
  k_build_inp<<<(Bsz*Ssz*XPW+255)/256,256,0,stream>>>(ic, wemb, cids, wids, inpw);

  // ---- word-level bi-LSTM over sentences -----------------------------------
  for (int dir = 0; dir < 2; dir++){
    k_zero_f16<<<(Bsz*KW+255)/256,256,0,stream>>>(xhw[0], Bsz*KW);
    k_zero_f16<<<(Bsz*KW+255)/256,256,0,stream>>>(xhw[1], Bsz*KW);
    k_zero_f32<<<(Bsz*Dd+255)/256,256,0,stream>>>(cword[dir], Bsz*Dd);
    k_zero_f32<<<(Bsz*Dd+255)/256,256,0,stream>>>(hword[dir], Bsz*Dd);
    k_zero_f32<<<(Bsz*Ssz*Dd+255)/256,256,0,stream>>>(outw[dir], Bsz*Ssz*Dd);
    for (int t = 0; t < Ssz; t++){
      k_word_x<<<(Bsz*XPW+255)/256,256,0,stream>>>(inpw, slens, xhw[t&1], t, dir);
      k_lstm_step<KW,KSW,4><<<Bsz/8,256,0,stream>>>(
          xhw[t&1], xhw[(t+1)&1], pkKw[dir], bwb[dir],
          cword[dir], hword[dir], outw[dir], slens, t, dir);
    }
  }

  // ---- attention + head -----------------------------------------------------
  for (int dir = 0; dir < 2; dir++)
    k_attn<<<Bsz,256,0,stream>>>(outw[dir], slens, hword[dir], Wm[dir], Wq[dir], vv[dir], ctx[dir]);
  k_head<<<Bsz,256,0,stream>>>(ctx[0], ctx[1], cword[0], hword[0], cword[1], hword[1],
                               W1, b1, W2, b2, (float*)d_out);
}
